// LcaModule_5162550690583
// MI455X (gfx1250) — matrix operand order repaired and checked by reading, not compiled, not run on hardware
//
#include <hip/hip_runtime.h>

// ---------------------------------------------------------------------------
// LCA sparse coding on MI455X (gfx1250, wave32, WMMA bf16 16x16x32)
//   b = x@w ; g = w^T w - I ; 49x { a=relu(u-l); u = 0.9u + 0.1b - 0.1 a@g }
//   out = relu(u-l) @ w^T
// Dominant cost: 49 bf16 4096^3 GEMMs -> v_wmma_f32_16x16x32_bf16, fed by
// async global->LDS copies (ASYNCcnt) with double-buffered LDS tiles.
// ---------------------------------------------------------------------------

#define B_DIM   4096   // batch rows (M)
#define L_DIM   4096   // latents
#define P_DIM   1024   // pixels
#define NSTEPS  49
#define LAMBDA  0.3f
#define DT      0.1f

typedef __attribute__((ext_vector_type(16))) __bf16 v16bf;
typedef __attribute__((ext_vector_type(8)))  float  v8f;

union FragBF {
  v16bf v;
  unsigned int u[8];
};

__device__ __forceinline__ unsigned short f32_to_bf16_rne(float f) {
  unsigned int x = __float_as_uint(f);
  x += 0x7FFFu + ((x >> 16) & 1u);
  return (unsigned short)(x >> 16);
}

// ------------------- gfx1250 async global->LDS machinery -------------------

#if __has_builtin(__builtin_amdgcn_global_load_async_to_lds_b128)
#define USE_ASYNC_LDS 1
#else
#define USE_ASYNC_LDS 0
#endif

#if USE_ASYNC_LDS
typedef int v4i __attribute__((vector_size(16)));
typedef __attribute__((address_space(1))) v4i as1_v4i;   // global
typedef __attribute__((address_space(3))) v4i as3_v4i;   // LDS

__device__ __forceinline__ void async_copy_b128(const unsigned short* g,
                                                unsigned short* lds) {
  // LDS generic address: low 32 bits are the LDS offset.
  __builtin_amdgcn_global_load_async_to_lds_b128(
      (as1_v4i*)(unsigned long long)g,
      (as3_v4i*)(unsigned)(unsigned long long)lds,
      0, 0);
}
#endif

__device__ __forceinline__ void wait_async0() {
#if USE_ASYNC_LDS
#if __has_builtin(__builtin_amdgcn_s_wait_asynccnt)
  __builtin_amdgcn_s_wait_asynccnt(0);
#else
  asm volatile("s_wait_asynccnt 0" ::: "memory");
#endif
#endif
}

// --------------------------- elementwise kernels ---------------------------

__global__ __launch_bounds__(256) void zero_f32(float* p, int n) {
  int i = blockIdx.x * 256 + threadIdx.x;
  if (i < n) p[i] = 0.0f;
}

// a_bf16 = bf16(relu(u - lambda))
__global__ __launch_bounds__(256) void lca_thresh(const float* __restrict__ u,
                                                  unsigned short* __restrict__ a,
                                                  int n) {
  int i = blockIdx.x * 256 + threadIdx.x;
  if (i < n) {
    float v = u[i] - LAMBDA;
    v = v > 0.0f ? v : 0.0f;
    a[i] = f32_to_bf16_rne(v);
  }
}

// x_bf16[i]=bf16(x[i]); w_bf16[i]=bf16(w[i]); wt_bf16[l][p]=bf16(w[p][l])
__global__ __launch_bounds__(256) void convert_inputs(const float* __restrict__ x,
                                                      const float* __restrict__ w,
                                                      unsigned short* __restrict__ xb,
                                                      unsigned short* __restrict__ wb,
                                                      unsigned short* __restrict__ wt,
                                                      int n) {
  int i = blockIdx.x * 256 + threadIdx.x;
  if (i < n) {
    xb[i] = f32_to_bf16_rne(x[i]);
    unsigned short wv = f32_to_bf16_rne(w[i]);
    wb[i] = wv;                       // w row-major [P][L]
    int p = i >> 12;                  // / L_DIM
    int l = i & (L_DIM - 1);
    wt[(size_t)l * P_DIM + p] = wv;   // w^T row-major [L][P]
  }
}

// ------------------------------ WMMA GEMM ----------------------------------
// C[M,N] = A[M,K](bf16 row-major) * Bt[N,K]^T(bf16, N-major)
// MODE 0: Cf = acc                          (f32 out: b, reconstruction)
// MODE 1: Cbf = bf16(acc - I)               (g = w^T w - I)
// MODE 2: Uf = 0.9*Uf + 0.1*Bsrc - 0.1*acc  (LCA step, in-place on u)
//
// Block: 128x128 C tile, 256 threads = 8 wave32 waves arranged 2(M) x 4(N);
// each wave: 4x2 grid of 16x16 tiles. K staged 32 deep through double-
// buffered LDS fed by global_load_async_to_lds_b128 (ASYNCcnt pipeline).

template <int MODE>
__global__ __launch_bounds__(256) void wmma_gemm(
    const unsigned short* __restrict__ A,
    const unsigned short* __restrict__ Bt,
    const float* __restrict__ Bsrc,
    float* __restrict__ Uf,
    float* __restrict__ Cf,
    unsigned short* __restrict__ Cbf,
    int M, int N, int K) {
  constexpr int BM = 128, BN = 128, BK = 32;
  constexpr int LDK = 40;  // padded row stride (80B): conflict-free, 16B-aligned
  __shared__ unsigned short sA[2][BM * LDK];
  __shared__ unsigned short sB[2][BN * LDK];

  const int tid   = threadIdx.x;
  const int lane  = tid & 31;
  const int wave  = tid >> 5;
  const int waveM = wave >> 2;      // 0..1  -> 64-row slab
  const int waveN = wave & 3;       // 0..3  -> 32-col slab
  const int half  = lane >> 4;      // hi/lo 16 lanes
  const int l16   = lane & 15;

  const int blockRow = blockIdx.y * BM;
  const int blockCol = blockIdx.x * BN;

  v8f acc[4][2] = {};

  // Staging: each thread moves 16 bf16 (two b128) per tile per matrix.
  const int ldRow = tid >> 1;          // 0..127
  const int ldCol = (tid & 1) << 4;    // 0 or 16
  const unsigned short* Ag = A  + (size_t)(blockRow + ldRow) * K + ldCol;
  const unsigned short* Bg = Bt + (size_t)(blockCol + ldRow) * K + ldCol;

  auto stage = [&](int bufi, int k0) {
    unsigned short* da = &sA[bufi][ldRow * LDK + ldCol];
    unsigned short* db = &sB[bufi][ldRow * LDK + ldCol];
    const unsigned short* ga = Ag + k0;
    const unsigned short* gb = Bg + k0;
#if USE_ASYNC_LDS
    async_copy_b128(ga,     da);
    async_copy_b128(ga + 8, da + 8);
    async_copy_b128(gb,     db);
    async_copy_b128(gb + 8, db + 8);
#else
    uint4 t0 = ((const uint4*)ga)[0];
    uint4 t1 = ((const uint4*)ga)[1];
    uint4 t2 = ((const uint4*)gb)[0];
    uint4 t3 = ((const uint4*)gb)[1];
    ((uint4*)da)[0] = t0;
    ((uint4*)da)[1] = t1;
    ((uint4*)db)[0] = t2;
    ((uint4*)db)[1] = t3;
#endif
  };

  // Prologue: stage tile 0 into buffer 0.
  stage(0, 0);
  wait_async0();
  __syncthreads();

  int buf = 0;
  for (int k0 = 0; k0 < K; k0 += BK) {
    // Kick off next tile into the other buffer while we compute.
    if (k0 + BK < K) {
      stage(buf ^ 1, k0 + BK);
      if (k0 + 2 * BK < K) {               // L2 prefetch two tiles ahead
        __builtin_prefetch(Ag + k0 + 2 * BK, 0, 1);
        __builtin_prefetch(Bg + k0 + 2 * BK, 0, 1);
      }
    }

    const unsigned short* bufA = sA[buf];
    const unsigned short* bufB = sB[buf];

    // B fragments: lane = N; lanes 0-15 hold K=0..15 (pairs), 16-31 hold K=16..31
    FragBF fb[2];
#pragma unroll
    for (int nI = 0; nI < 2; ++nI) {
      const unsigned short* row = bufB + (waveN * 32 + nI * 16 + l16) * LDK;
#pragma unroll
      for (int i = 0; i < 8; ++i) {
        const int kk = 2 * i + 16 * half;
        fb[nI].u[i] = *(const unsigned int*)(row + kk);
      }
    }

#pragma unroll
    for (int mI = 0; mI < 4; ++mI) {
      // A fragment: lane = M; lanes 0-15 hold K=0..15, 16-31 hold K=16..31
      FragBF fa;
      const unsigned short* row = bufA + (waveM * 64 + mI * 16 + l16) * LDK;
#pragma unroll
      for (int i = 0; i < 8; ++i) {
        const int kb = 2 * i + 16 * half;
        fa.u[i] = *(const unsigned int*)(row + kb);
      }
#pragma unroll
      for (int nI = 0; nI < 2; ++nI) {
        acc[mI][nI] = __builtin_amdgcn_wmma_f32_16x16x32_bf16(
            false, fa.v, false, fb[nI].v, (short)0, acc[mI][nI], false, false);
      }
    }

    wait_async0();    // next tile landed in LDS
    __syncthreads();  // all waves done reading current tile
    buf ^= 1;
  }

  // Epilogue. C layout: VGPR r -> M = r + 8*half, N = lane&15.
#pragma unroll
  for (int mI = 0; mI < 4; ++mI) {
#pragma unroll
    for (int nI = 0; nI < 2; ++nI) {
      const int col = blockCol + waveN * 32 + nI * 16 + l16;
#pragma unroll
      for (int r = 0; r < 8; ++r) {
        const int row = blockRow + waveM * 64 + mI * 16 + r + 8 * half;
        const size_t idx = (size_t)row * N + col;
        float v = acc[mI][nI][r];
        if constexpr (MODE == 0) {
          Cf[idx] = v;
        } else if constexpr (MODE == 1) {
          if (row == col) v -= 1.0f;
          Cbf[idx] = f32_to_bf16_rne(v);
        } else {
          const float uo = Uf[idx];
          Uf[idx] = (1.0f - DT) * uo + DT * Bsrc[idx] - DT * v;
        }
      }
    }
  }
}

// ------------------------------- driver ------------------------------------

extern "C" void kernel_launch(void* const* d_in, const int* in_sizes, int n_in,
                              void* d_out, int out_size, void* d_ws, size_t ws_size,
                              hipStream_t stream) {
  const float* x = (const float*)d_in[0];   // [4096,1024] f32
  const float* w = (const float*)d_in[1];   // [1024,4096] f32
  float* out = (float*)d_out;               // [4096,1024] f32

  char* ws = (char*)d_ws;
  float* b = (float*)ws;               ws += (size_t)B_DIM * L_DIM * 4;  //  64 MB
  float* u = (float*)ws;               ws += (size_t)B_DIM * L_DIM * 4;  //  64 MB
  unsigned short* g  = (unsigned short*)ws; ws += (size_t)L_DIM * L_DIM * 2; // 32 MB
  unsigned short* a  = (unsigned short*)ws; ws += (size_t)B_DIM * L_DIM * 2; // 32 MB
  unsigned short* xb = (unsigned short*)ws; ws += (size_t)B_DIM * P_DIM * 2; //  8 MB
  unsigned short* wt = (unsigned short*)ws; ws += (size_t)L_DIM * P_DIM * 2; //  8 MB (w^T)
  unsigned short* wb = (unsigned short*)ws; ws += (size_t)P_DIM * L_DIM * 2; //  8 MB (w)

  const int nXW = B_DIM * P_DIM;   // 4M (== P*L)
  const int nUL = B_DIM * L_DIM;   // 16M

  convert_inputs<<<(nXW + 255) / 256, 256, 0, stream>>>(x, w, xb, wb, wt, nXW);
  zero_f32<<<(nUL + 255) / 256, 256, 0, stream>>>(u, nUL);

  dim3 blk(256);
  // b = x @ w            : A=xb[B,P], Bt=wt[L,P]
  wmma_gemm<0><<<dim3(L_DIM / 128, B_DIM / 128), blk, 0, stream>>>(
      xb, wt, nullptr, nullptr, b, nullptr, B_DIM, L_DIM, P_DIM);
  // g = w^T w - I        : A=wt[L,P], Bt=wt[L,P]
  wmma_gemm<1><<<dim3(L_DIM / 128, L_DIM / 128), blk, 0, stream>>>(
      wt, wt, nullptr, nullptr, nullptr, g, L_DIM, L_DIM, P_DIM);

  for (int s = 0; s < NSTEPS; ++s) {
    lca_thresh<<<(nUL + 255) / 256, 256, 0, stream>>>(u, a, nUL);
    // u = 0.9u + 0.1b - 0.1 * (a @ g)   (g symmetric -> g is its own Bt)
    wmma_gemm<2><<<dim3(L_DIM / 128, B_DIM / 128), blk, 0, stream>>>(
        a, g, b, u, nullptr, nullptr, B_DIM, L_DIM, L_DIM);
  }

  lca_thresh<<<(nUL + 255) / 256, 256, 0, stream>>>(u, a, nUL);
  // out = a @ w^T        : A=a[B,L], Bt=wb[P,L]
  wmma_gemm<0><<<dim3(P_DIM / 128, B_DIM / 128), blk, 0, stream>>>(
      a, wb, nullptr, nullptr, out, nullptr, B_DIM, P_DIM, L_DIM);
}
